// CPPN_14568529068326
// MI455X (gfx1250) — matrix operand order repaired and checked by reading, not compiled, not run on hardware
//
#include <hip/hip_runtime.h>

#define NPTS    131072
#define IN_DIM  12
#define HID     256
#define NLAYERS 10
#define OUT_DIM 3
#define ROWS    128     // batch rows per workgroup
#define HSTR    264     // LDS row stride (bf16 elems) for activations
#define WSTR    264     // LDS row stride (bf16 elems) for staged weights

typedef __bf16 bf16;
typedef __attribute__((ext_vector_type(8)))  bf16  bf16x8;
typedef __attribute__((ext_vector_type(16))) bf16  bf16x16;
typedef __attribute__((ext_vector_type(8)))  float v8f;

union Frag { bf16x16 v; bf16x8 h[2]; };

// ---------------------------------------------------------------------------
// Prep: hidden weights f32 -> bf16, transposed: Wt[layer][n][k] = Ws[layer][k][n]
// ---------------------------------------------------------------------------
__global__ __launch_bounds__(256) void cvt_weights(const float* __restrict__ Ws,
                                                   bf16* __restrict__ Wt) {
  int idx = blockIdx.x * 256 + threadIdx.x;
  if (idx >= (NLAYERS - 1) * HID * HID) return;
  int l = idx >> 16;
  int r = idx & 65535;
  int k = r >> 8;
  int n = r & 255;
  Wt[l * (HID * HID) + n * HID + k] = (bf16)Ws[idx];
}

__device__ __forceinline__ float actf(float h, int m) {
  switch (m & 3) {
    case 0:  return __sinf(h);
    case 1:  return __cosf(h);
    case 2:  return __expf(-h * h);
    default: return tanhf(h);
  }
}

// CDNA5 async copy: 16B global -> LDS, tracked by ASYNCcnt.
__device__ __forceinline__ void async_cp16(const bf16* g, const bf16* l) {
  unsigned loff = (unsigned)(unsigned long long)(const void*)l;  // flat LDS addr low 32b = LDS offset
  asm volatile("global_load_async_to_lds_b128 %0, %1, off"
               :: "v"(loff), "v"((unsigned long long)g) : "memory");
}
__device__ __forceinline__ void wait_async0() {
  asm volatile("s_wait_asynccnt 0" ::: "memory");
}

// Stage one 256x256 bf16 weight matrix (row-major [n][k]) into padded LDS.
__device__ __forceinline__ void stage_weights(const bf16* __restrict__ Wsrc,
                                              bf16* __restrict__ Wl, int tid) {
#pragma unroll
  for (int i = 0; i < 32; ++i) {          // 8192 16B-chunks / 256 threads
    int idx = i * 256 + tid;
    int r = idx >> 5;                     // row (n), 32 chunks per row
    int c = idx & 31;                     // 16B chunk within row
    async_cp16(Wsrc + r * HID + c * 8, Wl + r * WSTR + c * 8);
  }
}

// ---------------------------------------------------------------------------
// Main CPPN kernel: LDS-staged weights, 64x64 register-blocked WMMA per wave.
// ---------------------------------------------------------------------------
__global__ __launch_bounds__(256) void cppn_kernel(
    const float* __restrict__ x,    const float* __restrict__ W0,
    const float* __restrict__ b0,   const bf16*  __restrict__ Wt,
    const float* __restrict__ bs,   const float* __restrict__ Wout,
    const float* __restrict__ bout, float* __restrict__ out) {
  __shared__ alignas(16) bf16 H[ROWS][HSTR];       // activations, 67.5 KB
  __shared__ alignas(16) bf16 Wlds[HID][WSTR];     // staged weights, 135 KB
  __shared__ float xs[ROWS * IN_DIM];              // input tile, 6 KB

  const int tid  = threadIdx.x;
  const int row0 = blockIdx.x * ROWS;

  // Kick off async staging of layer-1 weights immediately.
  stage_weights(Wt, &Wlds[0][0], tid);

  // Stage input tile
  for (int i = tid; i < ROWS * IN_DIM; i += 256)
    xs[i] = x[row0 * IN_DIM + i];
  __syncthreads();

  // Layer 0: K=12, scalar f32 (trivial FLOP share), act = sin
  for (int i = tid; i < ROWS * HID; i += 256) {
    int r = i >> 8, c = i & 255;
    float a = b0[c];
#pragma unroll
    for (int k = 0; k < IN_DIM; ++k)
      a += xs[r * IN_DIM + k] * W0[k * HID + c];
    H[r][c] = (bf16)__sinf(a);
  }
  wait_async0();          // layer-1 weights landed in LDS
  __syncthreads();        // H + Wlds visible to all waves

  const int lane  = tid & 31;
  const int wave  = tid >> 5;
  const int nlo   = lane & 15;
  const int khalf = lane >> 4;
  const int mg    = (wave >> 2) * 64;   // wave's 64-row group (2 groups)
  const int ng    = (wave & 3)  * 64;   // wave's 64-col group (4 groups)

#pragma unroll 1
  for (int layer = 1; layer < NLAYERS; ++layer) {
    const float* bl = bs + (layer - 1) * HID;

    v8f acc[4][4] = {};
    for (int s = 0; s < 8; ++s) {       // K steps of 32
      const int k0 = s * 32;
      Frag A[4], B[4];
#pragma unroll
      for (int mt = 0; mt < 4; ++mt) {  // A: 16x32 bf16 fragments from H
        const bf16* ap = &H[mg + mt * 16 + nlo][k0 + khalf * 8];
        A[mt].h[0] = *(const bf16x8*)ap;
        A[mt].h[1] = *(const bf16x8*)(ap + 16);
      }
#pragma unroll
      for (int nt = 0; nt < 4; ++nt) {  // B: 32x16 bf16 fragments from Wlds
        const bf16* bp = &Wlds[ng + nt * 16 + nlo][k0 + khalf * 8];
        B[nt].h[0] = *(const bf16x8*)bp;
        B[nt].h[1] = *(const bf16x8*)(bp + 16);
      }
#pragma unroll
      for (int mt = 0; mt < 4; ++mt)
#pragma unroll
        for (int nt = 0; nt < 4; ++nt)
          acc[mt][nt] = __builtin_amdgcn_wmma_f32_16x16x32_bf16(
              false, A[mt].v, false, B[nt].v, (short)0, acc[mt][nt], false, false);
    }

    __syncthreads();      // all H / Wlds reads of this layer complete

    // Overlap: start DMA of next layer's weights while doing the epilogue.
    if (layer < NLAYERS - 1)
      stage_weights(Wt + layer * HID * HID, &Wlds[0][0], tid);

    // Epilogue: bias + activation (f32 VALU), C layout -> row-major bf16 in H
#pragma unroll
    for (int mt = 0; mt < 4; ++mt)
#pragma unroll
      for (int nt = 0; nt < 4; ++nt) {
        const float bias = bl[ng + nt * 16 + nlo];
#pragma unroll
        for (int r = 0; r < 8; ++r) {
          float h = acc[mt][nt][r] + bias;
          H[mg + mt * 16 + (khalf ? r + 8 : r)][ng + nt * 16 + nlo] =
              (bf16)actf(h, layer);
        }
      }

    wait_async0();        // next layer's weights landed
    __syncthreads();      // epilogue H writes + Wlds visible
  }

  // Output head: 256 -> 3, sigmoid, scalar f32 (trivial FLOP share)
  for (int i = tid; i < ROWS * OUT_DIM; i += 256) {
    int r = i / OUT_DIM, o = i - r * OUT_DIM;
    float a = bout[o];
    for (int k = 0; k < HID; ++k)
      a += (float)H[r][k] * Wout[k * OUT_DIM + o];
    out[(row0 + r) * OUT_DIM + o] = 1.0f / (1.0f + __expf(-a));
  }
}

// ---------------------------------------------------------------------------
extern "C" void kernel_launch(void* const* d_in, const int* in_sizes, int n_in,
                              void* d_out, int out_size, void* d_ws, size_t ws_size,
                              hipStream_t stream) {
  const float* x    = (const float*)d_in[0];
  const float* W0   = (const float*)d_in[1];
  const float* b0   = (const float*)d_in[2];
  const float* Ws   = (const float*)d_in[3];
  const float* bs   = (const float*)d_in[4];
  const float* Wout = (const float*)d_in[5];
  const float* bout = (const float*)d_in[6];
  float* out = (float*)d_out;
  bf16* Wt   = (bf16*)d_ws;   // (NLAYERS-1)*256*256 bf16 = ~1.2 MB

  const int wtot = (NLAYERS - 1) * HID * HID;
  cvt_weights<<<(wtot + 255) / 256, 256, 0, stream>>>(Ws, Wt);
  cppn_kernel<<<NPTS / ROWS, 256, 0, stream>>>(x, W0, b0, Wt, bs, Wout, bout, out);
}
